// TrajectoryLSTM_33938831573145
// MI455X (gfx1250) — matrix operand order repaired and checked by reading, not compiled, not run on hardware
//
#include <hip/hip_runtime.h>

// ---------------- problem constants ----------------
#define Bsz   16384
#define Tt    4
#define Ff    7
#define Ee    64
#define Hh    128
#define NHh   4
#define HDd   32
#define Nn    32
#define Kk    3

typedef __bf16 bf16_t;
typedef __bf16 v16bf __attribute__((ext_vector_type(16)));
typedef float  v8f   __attribute__((ext_vector_type(8)));

__device__ inline bf16_t f2bf(float f) { return (bf16_t)f; }   // native v_cvt
__device__ inline float  bf2f(bf16_t b) { return (float)b; }
__device__ inline float  rcpf(float x) { return __builtin_amdgcn_rcpf(x); }
// branch-free: sigmoid / tanh via hardware exp + rcp (exact at +/-inf)
__device__ inline float sigmf(float x) { return rcpf(1.0f + __expf(-x)); }
__device__ inline float tanh_fast(float x) {
  float e = __expf(2.0f * x);
  return 1.0f - 2.0f * rcpf(e + 1.0f);
}

__device__ inline v8f vzero8() {
  v8f z = {0.f, 0.f, 0.f, 0.f, 0.f, 0.f, 0.f, 0.f};
  return z;
}

// A fragment: 16x32 bf16 tile from LDS, row-major, row stride S (elements).
// A order used here (contiguous K, not the ISA table order): lane m = lane&15, half = lane>>4;
// VGPR j holds K = 2*j + 16*half, +1.
__device__ inline v16bf load_a(const bf16_t* base, int S, int lane) {
  const bf16_t* row = base + (lane & 15) * S;
  const int h8 = (lane >> 4) << 4;
  v16bf a;
#pragma unroll
  for (int j = 0; j < 8; ++j) {
    int k = (j << 1) + h8;
    a[2 * j]     = row[k];
    a[2 * j + 1] = row[k + 1];
  }
  return a;
}

// B fragment: weights stored (out n, in k) row-major = B^T, stride S.
// B order used here (contiguous K, same as load_a): lane n = lane&15, element e holds K = k0 + 16*half + e.
// Unconditional float4 loads (64B-aligned) -> global_load_b128, then native
// fp32->bf16 converts. No guards: callers guarantee n0+15 < rows.
__device__ inline v16bf load_b_f32(const float* W, int n0, int k0, int S,
                                   int lane) {
  const float* row =
      W + (size_t)(n0 + (lane & 15)) * S + k0 + ((lane >> 4) << 4);
  v16bf b;
#pragma unroll
  for (int q = 0; q < 4; ++q) {
    float4 f = ((const float4*)row)[q];
    b[4 * q + 0] = f2bf(f.x);
    b[4 * q + 1] = f2bf(f.y);
    b[4 * q + 2] = f2bf(f.z);
    b[4 * q + 3] = f2bf(f.w);
  }
  return b;
}

// Guarded variant for padded matrices (traj_W2: 36 rows -> 48-row tile).
// Row index is CLAMPED so the load itself is unconditional & vectorized;
// out-of-range rows are zeroed by a post-load multiply with a 0/1 factor.
__device__ inline v16bf load_b_f32_pad(const float* W, int n0, int k0, int S,
                                       int nmax, int lane) {
  const int n = n0 + (lane & 15);
  const int nc = (n < nmax) ? n : (nmax - 1);
  const float zf = (n < nmax) ? 1.0f : 0.0f;
  const float* row = W + (size_t)nc * S + k0 + ((lane >> 4) << 4);
  v16bf b;
#pragma unroll
  for (int q = 0; q < 4; ++q) {
    float4 f = ((const float4*)row)[q];
    b[4 * q + 0] = f2bf(f.x * zf);
    b[4 * q + 1] = f2bf(f.y * zf);
    b[4 * q + 2] = f2bf(f.z * zf);
    b[4 * q + 3] = f2bf(f.w * zf);
  }
  return b;
}

__device__ inline v8f wmma_bf16(v16bf a, v16bf b, v8f c) {
  return __builtin_amdgcn_wmma_f32_16x16x32_bf16(
      false, a, false, b, (short)0, c, false, false);
}

// ------------------------------------------------------------------
// Fused trajectory model: one WG = 16 batch rows, 256 threads (8 waves).
// ------------------------------------------------------------------
__global__ __launch_bounds__(256) void traj_fused(
    const float* __restrict__ x, const float* __restrict__ nbrs,
    const int* __restrict__ ncnt,
    const float* __restrict__ embW, const float* __restrict__ embB,
    const float* __restrict__ Wih, const float* __restrict__ Whh,
    const float* __restrict__ bih, const float* __restrict__ bhh,
    const float* __restrict__ inW, const float* __restrict__ inB,
    const float* __restrict__ outW, const float* __restrict__ outB,
    const float* __restrict__ tW1, const float* __restrict__ tB1,
    const float* __restrict__ tW2, const float* __restrict__ tB2,
    const float* __restrict__ pW, const float* __restrict__ pB,
    float* __restrict__ out_traj, float* __restrict__ out_prob,
    float* __restrict__ out_attn) {
  __shared__ float  s_ht[16 * Hh];    // target-encoder h_t, fp32 (persist)
  __shared__ float  s_hs[16 * Hh];    // h_social, fp32 (persist)
  __shared__ bf16_t s_hc[16 * 256];   // [h_t | h_social] bf16 (persist)
  __shared__ __align__(16) char s_arena[32 * 1024];   // phase scratch

  const int tid  = threadIdx.x;
  const int lane = tid & 31;
  const int wave = tid >> 5;
  const int r0   = blockIdx.x * 16;

  // ================= Phase A: target encoder =================
  {
    float*  s_x    = (float*)s_arena;                        // [16][4][8]
    bf16_t* s_embt = (bf16_t*)(s_arena + 2048);              // [4][16*64]
    bf16_t* s_h    = (bf16_t*)(s_arena + 2048 + 8192);       // [16*128]
    float*  s_c    = (float*)(s_arena + 2048 + 8192 + 4096); // [16*128]

    for (int i = tid; i < 16 * Tt * Ff; i += 256) {
      int m = i / (Tt * Ff), rem = i % (Tt * Ff);
      s_x[(m * Tt + rem / Ff) * 8 + (rem % Ff)] =
          x[(size_t)(r0 + m) * Tt * Ff + rem];
    }
    for (int i = tid; i < 16 * Hh; i += 256) {
      s_h[i] = f2bf(0.0f);
      s_c[i] = 0.0f;
    }
    __syncthreads();

    // embed + relu, bf16, per timestep
    for (int i = tid; i < Tt * 16 * Ee; i += 256) {
      int t = i / (16 * Ee), rem = i % (16 * Ee);
      int m = rem >> 6, e = rem & 63;
      float acc = embB[e];
      const float* xr = &s_x[(m * Tt + t) * 8];
      const float* wr = &embW[e * Ff];
#pragma unroll
      for (int f = 0; f < Ff; ++f) acc += xr[f] * wr[f];
      s_embt[t * 16 * Ee + rem] = f2bf(fmaxf(acc, 0.0f));
    }
    __syncthreads();

    // 4 LSTM steps: wave owns h-columns [wave*16, wave*16+16), all 4 gates.
    const int j    = wave;
    const int hcol = j * 16 + (lane & 15);
    float bsum[4];
#pragma unroll
    for (int g = 0; g < 4; ++g)
      bsum[g] = bih[g * Hh + hcol] + bhh[g * Hh + hcol];

    // Hoist the recurrent-weight fragments (invariant over the 4 steps).
    v16bf bWhh[4][4];
#pragma unroll
    for (int g = 0; g < 4; ++g)
#pragma unroll
      for (int ks = 0; ks < 4; ++ks)
        bWhh[g][ks] = load_b_f32(Whh, g * Hh + j * 16, ks * 32, Hh, lane);

    for (int t = 0; t < Tt; ++t) {
      v8f acc[4];
#pragma unroll
      for (int g = 0; g < 4; ++g) acc[g] = vzero8();

      v16bf aemb[2], ah[4];
      aemb[0] = load_a(s_embt + t * 16 * Ee + 0, Ee, lane);
      aemb[1] = load_a(s_embt + t * 16 * Ee + 32, Ee, lane);
#pragma unroll
      for (int ks = 0; ks < 4; ++ks) ah[ks] = load_a(s_h + ks * 32, Hh, lane);

#pragma unroll
      for (int g = 0; g < 4; ++g) {
        const int n0 = g * Hh + j * 16;
#pragma unroll
        for (int ks = 0; ks < 2; ++ks)
          acc[g] = wmma_bf16(aemb[ks],
                             load_b_f32(Wih, n0, ks * 32, Ee, lane), acc[g]);
#pragma unroll
        for (int ks = 0; ks < 4; ++ks)
          acc[g] = wmma_bf16(ah[ks], bWhh[g][ks], acc[g]);
      }
      __syncthreads();   // all waves done reading s_h before we overwrite it

#pragma unroll
      for (int r = 0; r < 8; ++r) {
        const int m = r + ((lane >> 4) << 3);
        float iv = sigmf(acc[0][r] + bsum[0]);
        float fv = sigmf(acc[1][r] + bsum[1]);
        float gv = tanh_fast(acc[2][r] + bsum[2]);
        float ov = sigmf(acc[3][r] + bsum[3]);
        float c  = fv * s_c[m * Hh + hcol] + iv * gv;
        float h  = ov * tanh_fast(c);
        s_c[m * Hh + hcol] = c;
        s_h[m * Hh + hcol] = f2bf(h);
        if (t == Tt - 1) {
          s_ht[m * Hh + hcol]  = h;
          s_hc[m * 256 + hcol] = f2bf(h);
        }
      }
      __syncthreads();
    }
  }

  // ================= Phase B: neighbor encoder + social attention =========
  {
    float*  a_nbr = (float*)s_arena;                               // 32*8
    float*  a_pp  = (float*)s_arena;                 // 256 partials (aliases
                                                     // a_nbr; dead by then)
    bf16_t* a_emb = (bf16_t*)(s_arena + 1024);                     // 32*64
    bf16_t* a_hn  = (bf16_t*)(s_arena + 1024 + 4096);              // 32*128
    bf16_t* a_kv  = (bf16_t*)(s_arena + 1024 + 4096 + 8192);       // 32*256
    float*  a_q   = (float*)(s_arena + 1024 + 4096 + 8192 + 16384);
    float*  a_sc  = a_q + 128;
    float*  a_w   = a_sc + 128;
    float*  a_at  = a_w + 128;

    // ---- hoisted, row-invariant weight fragments (kept in VGPRs) ----
    // gates: wave owns h-slice jn = wave (i,g,o only; f-gate dead: c0 = 0)
    const int jn = wave;
    const int hc = jn * 16 + (lane & 15);
    v16bf bg_i[2], bg_g[2], bg_o[2];
#pragma unroll
    for (int ks = 0; ks < 2; ++ks) {
      bg_i[ks] = load_b_f32(Wih, 0 * Hh + jn * 16, ks * 32, Ee, lane);
      bg_g[ks] = load_b_f32(Wih, 2 * Hh + jn * 16, ks * 32, Ee, lane);
      bg_o[ks] = load_b_f32(Wih, 3 * Hh + jn * 16, ks * 32, Ee, lane);
    }
    const float bI = bih[hc] + bhh[hc];
    const float bG = bih[2 * Hh + hc] + bhh[2 * Hh + hc];
    const float bO = bih[3 * Hh + hc] + bhh[3 * Hh + hc];
    // K/V: wave owns n-tiles {2*wave, 2*wave+1} of the 256 output columns
    v16bf bkv[2][4];
    float bias_kv[2];
#pragma unroll
    for (int t2 = 0; t2 < 2; ++t2) {
      const int n0 = (wave * 2 + t2) * 16;
#pragma unroll
      for (int ks = 0; ks < 4; ++ks)
        bkv[t2][ks] = load_b_f32(inW, Hh + n0, ks * 32, Hh, lane);
      bias_kv[t2] = inB[Hh + n0 + (lane & 15)];
    }

    for (int bi = 0; bi < 16; ++bi) {
      const int b   = r0 + bi;
      const int cnt = ncnt[b];

      // prefetch next row's neighbor block (global_prefetch_b8)
      if (bi + 1 < 16 && tid < Nn)
        __builtin_prefetch(&nbrs[(size_t)(b + 1) * Nn * Ff + tid * Ff], 0, 0);

      for (int i = tid; i < Nn * Ff; i += 256)
        a_nbr[(i / Ff) * 8 + (i % Ff)] = nbrs[(size_t)b * Nn * Ff + i];
      __syncthreads();

      // embed + relu for 32 neighbors
      for (int i = tid; i < Nn * Ee; i += 256) {
        int m = i >> 6, e = i & 63;
        float acc = embB[e];
#pragma unroll
        for (int f = 0; f < Ff; ++f) acc += a_nbr[m * 8 + f] * embW[e * Ff + f];
        a_emb[i] = f2bf(fmaxf(acc, 0.0f));
      }
      __syncthreads();

      // gate GEMM: wave computes i/g/o for its 16 h-columns, both m-tiles
      {
#pragma unroll
        for (int mt = 0; mt < 2; ++mt) {
          const int m0 = mt * 16;
          v16bf a0 = load_a(a_emb + m0 * Ee + 0, Ee, lane);
          v16bf a1 = load_a(a_emb + m0 * Ee + 32, Ee, lane);
          v8f ai = vzero8(), ag = vzero8(), ao = vzero8();
          ai = wmma_bf16(a0, bg_i[0], ai);
          ai = wmma_bf16(a1, bg_i[1], ai);
          ag = wmma_bf16(a0, bg_g[0], ag);
          ag = wmma_bf16(a1, bg_g[1], ag);
          ao = wmma_bf16(a0, bg_o[0], ao);
          ao = wmma_bf16(a1, bg_o[1], ao);
#pragma unroll
          for (int r = 0; r < 8; ++r) {
            const int m = m0 + r + ((lane >> 4) << 3);
            float iv = sigmf(ai[r] + bI);
            float gv = tanh_fast(ag[r] + bG);
            float ov = sigmf(ao[r] + bO);
            a_hn[m * Hh + hc] = f2bf(ov * tanh_fast(iv * gv));
          }
        }
      }
      __syncthreads();

      // K/V projection GEMM: (32,128) @ (128,256)
      {
        v16bf ah[2][4];
#pragma unroll
        for (int mt = 0; mt < 2; ++mt)
#pragma unroll
          for (int ks = 0; ks < 4; ++ks)
            ah[mt][ks] = load_a(a_hn + mt * 16 * Hh + ks * 32, Hh, lane);
#pragma unroll
        for (int t2 = 0; t2 < 2; ++t2) {
          const int n0 = (wave * 2 + t2) * 16;
          const int col = n0 + (lane & 15);
#pragma unroll
          for (int mt = 0; mt < 2; ++mt) {
            v8f acc = vzero8();
#pragma unroll
            for (int ks = 0; ks < 4; ++ks)
              acc = wmma_bf16(ah[mt][ks], bkv[t2][ks], acc);
#pragma unroll
            for (int r = 0; r < 8; ++r) {
              const int m = mt * 16 + r + ((lane >> 4) << 3);
              a_kv[m * 256 + col] = f2bf(acc[r] + bias_kv[t2]);
            }
          }
        }
      }

      // q = h_t @ Wq^T + bq — split-K over all 256 threads
      {
        const int h = tid & 127, part = tid >> 7;
        const float* wr = &inW[(size_t)h * Hh + part * 64];
        const float* hr = &s_ht[bi * Hh + part * 64];
        float acc = 0.0f;
#pragma unroll
        for (int k2 = 0; k2 < 64; ++k2) acc += hr[k2] * wr[k2];
        a_pp[tid] = acc;
      }
      __syncthreads();
      if (tid < Hh) a_q[tid] = inB[tid] + a_pp[tid] + a_pp[tid + 128];
      __syncthreads();

      // scores (4 heads x 32 neighbors)
      if (tid < NHh * Nn) {
        const int head = tid >> 5, n = tid & 31;
        const int d0 = head * HDd;
        float s = 0.0f;
#pragma unroll
        for (int d = 0; d < HDd; ++d)
          s += a_q[d0 + d] * bf2f(a_kv[n * 256 + d0 + d]);
        s *= 0.17677669529663687f;   // 1/sqrt(32)
        if (n >= cnt) s = -1e9f;
        a_sc[tid] = s;
      }
      __syncthreads();

      // softmax per head (serial over 32, 4 threads)
      if (tid < NHh) {
        float mx = -1e30f;
        for (int n = 0; n < Nn; ++n) mx = fmaxf(mx, a_sc[tid * Nn + n]);
        float ssum = 0.0f;
        for (int n = 0; n < Nn; ++n) ssum += __expf(a_sc[tid * Nn + n] - mx);
        const float inv = rcpf(ssum);
        for (int n = 0; n < Nn; ++n)
          a_w[tid * Nn + n] = __expf(a_sc[tid * Nn + n] - mx) * inv;
      }
      __syncthreads();

      // attn = sum_n w * v; head-averaged attn weights out
      if (tid < Hh) {
        const int head = tid >> 5, d = tid & 31;
        float s = 0.0f;
        for (int n = 0; n < Nn; ++n)
          s += a_w[head * Nn + n] * bf2f(a_kv[n * 256 + Hh + head * HDd + d]);
        a_at[tid] = s;
      }
      if (tid < Nn) {
        float aw = 0.25f * (a_w[tid] + a_w[Nn + tid] + a_w[2 * Nn + tid] +
                            a_w[3 * Nn + tid]);
        out_attn[(size_t)b * Nn + tid] = (tid < cnt) ? aw : 0.0f;
      }
      __syncthreads();

      // out_proj — split-K over all 256 threads; zero if no neighbors
      {
        const int h = tid & 127, part = tid >> 7;
        const float* wr = &outW[(size_t)h * Hh + part * 64];
        float acc = 0.0f;
#pragma unroll
        for (int k2 = 0; k2 < 64; ++k2) acc += a_at[part * 64 + k2] * wr[k2];
        a_pp[tid] = acc;
      }
      __syncthreads();
      if (tid < Hh) {
        float acc = outB[tid] + a_pp[tid] + a_pp[tid + 128];
        if (cnt == 0) acc = 0.0f;
        s_hs[bi * Hh + tid]       = acc;
        s_hc[bi * 256 + Hh + tid] = f2bf(acc);
      }
      __syncthreads();
    }
  }

  // ================= Phase C: heads =================
  {
    bf16_t* c_hid = (bf16_t*)s_arena;                 // 16*128 bf16
    float*  c_lg  = (float*)(s_arena + 4096);         // 16*3 logits

    // GEMM1: (16,256)@(256,128) relu — wave owns n-tile = wave
    {
      const int n0 = wave * 16;
      v8f acc = vzero8();
#pragma unroll
      for (int ks = 0; ks < 8; ++ks)
        acc = wmma_bf16(load_a(s_hc + ks * 32, 256, lane),
                        load_b_f32(tW1, n0, ks * 32, 256, lane), acc);
      const int col = n0 + (lane & 15);
      const float bias = tB1[col];
#pragma unroll
      for (int r = 0; r < 8; ++r) {
        const int m = r + ((lane >> 4) << 3);
        c_hid[m * Hh + col] = f2bf(fmaxf(acc[r] + bias, 0.0f));
      }
    }
    __syncthreads();

    // GEMM2: (16,128)@(128,36→48pad); waves 0..2 (wave-uniform branch)
    if (wave < 3) {
      const int n0 = wave * 16;
      v8f acc = vzero8();
#pragma unroll
      for (int ks = 0; ks < 4; ++ks)
        acc = wmma_bf16(load_a(c_hid + ks * 32, Hh, lane),
                        load_b_f32_pad(tW2, n0, ks * 32, Hh, 36, lane), acc);
      const int col = n0 + (lane & 15);
      if (col < 36) {
        const float bias = tB2[col];
#pragma unroll
        for (int r = 0; r < 8; ++r) {
          const int m = r + ((lane >> 4) << 3);
          out_traj[(size_t)(r0 + m) * 36 + col] = acc[r] + bias;
        }
      }
    }

    // prob logits (tiny) + softmax over K=3
    if (tid < 16 * Kk) {
      const int m = tid / Kk, kk = tid % Kk;
      float acc = pB[kk];
      const float* wr = &pW[(size_t)kk * 256];
      for (int h2 = 0; h2 < Hh; ++h2)
        acc += s_ht[m * Hh + h2] * wr[h2] + s_hs[m * Hh + h2] * wr[Hh + h2];
      c_lg[tid] = acc;
    }
    __syncthreads();
    if (tid < 16) {
      const float l0 = c_lg[tid * 3], l1 = c_lg[tid * 3 + 1], l2 = c_lg[tid * 3 + 2];
      const float mx = fmaxf(l0, fmaxf(l1, l2));
      const float e0 = __expf(l0 - mx), e1 = __expf(l1 - mx), e2 = __expf(l2 - mx);
      const float inv = rcpf(e0 + e1 + e2);
      const size_t base = (size_t)(r0 + tid) * 3;
      out_prob[base]     = e0 * inv;
      out_prob[base + 1] = e1 * inv;
      out_prob[base + 2] = e2 * inv;
    }
  }
}

extern "C" void kernel_launch(void* const* d_in, const int* in_sizes, int n_in,
                              void* d_out, int out_size, void* d_ws,
                              size_t ws_size, hipStream_t stream) {
  (void)in_sizes; (void)n_in; (void)d_ws; (void)ws_size; (void)out_size;
  const float* x    = (const float*)d_in[0];
  const float* nbrs = (const float*)d_in[1];
  const int*   cnt  = (const int*)d_in[2];
  const float* embW = (const float*)d_in[3];
  const float* embB = (const float*)d_in[4];
  const float* Wih  = (const float*)d_in[5];
  const float* Whh  = (const float*)d_in[6];
  const float* bih  = (const float*)d_in[7];
  const float* bhh  = (const float*)d_in[8];
  const float* inW  = (const float*)d_in[9];
  const float* inB  = (const float*)d_in[10];
  const float* outW = (const float*)d_in[11];
  const float* outB = (const float*)d_in[12];
  const float* tW1  = (const float*)d_in[13];
  const float* tB1  = (const float*)d_in[14];
  const float* tW2  = (const float*)d_in[15];
  const float* tB2  = (const float*)d_in[16];
  const float* pW   = (const float*)d_in[17];
  const float* pB   = (const float*)d_in[18];

  float* out      = (float*)d_out;
  float* out_traj = out;                               // B*36
  float* out_prob = out + (size_t)Bsz * 36;            // B*3
  float* out_attn = out + (size_t)Bsz * 39;            // B*32

  traj_fused<<<Bsz / 16, 256, 0, stream>>>(
      x, nbrs, cnt, embW, embB, Wih, Whh, bih, bhh, inW, inB, outW, outB,
      tW1, tB1, tW2, tB2, pW, pB, out_traj, out_prob, out_attn);
}
